// RNN_46883863003344
// MI455X (gfx1250) — matrix operand order repaired and checked by reading, not compiled, not run on hardware
//
#include <hip/hip_runtime.h>
#include <hip/hip_bf16.h>
#include <math.h>

// Elman RNN for MI455X (gfx1250, wave32, WMMA).
//   S=512, B=256, I=512, H=512.
// Kernel 1: convert Vw, Ww f32 -> bf16 (workspace).
// Kernel 2: xV = x @ Vw^T + Vb  (bf16 WMMA GEMM, memory-bound, x read once).
// Kernel 3: persistent recurrence. 16 workgroups (one per 16 batch rows),
//           1024 threads = 32 waves on one WGP. Ww fragments resident in
//           VGPRs (16 N-tiles x 2 K-halves), h ping-pong in LDS, s_barrier
//           per step -> no cross-WGP sync on the 512-step critical path.
//           xV is register-prefetched one step ahead; the K-split partial
//           exchange is symmetric so both wave groups share the epilogue.

typedef __attribute__((ext_vector_type(8)))  __bf16 v8bf;
typedef __attribute__((ext_vector_type(16))) __bf16 v16bf;
typedef __attribute__((ext_vector_type(8)))  float  v8f;

__device__ __forceinline__ v16bf bf_cat(v8bf a, v8bf b) {
  return __builtin_shufflevector(a, b, 0,1,2,3,4,5,6,7,8,9,10,11,12,13,14,15);
}

// Load a 16x32 bf16 A/B-style fragment half-pair from bf16 memory.
// p points at this lane's first 8-half chunk; second chunk at p+step halves.
__device__ __forceinline__ v16bf load_frag(const __bf16* p, int step) {
  v8bf a = *(const v8bf*)p;
  v8bf b = *(const v8bf*)(p + step);
  return bf_cat(a, b);
}

// Convert 16 f32 (A-fragment source) to a bf16 fragment.
// p = lane's half0 start (8 floats); half1 at p+16 floats.
__device__ __forceinline__ v16bf cvtA(const float* p) {
  float4 a0 = *(const float4*)(p);
  float4 a1 = *(const float4*)(p + 4);
  float4 b0 = *(const float4*)(p + 16);
  float4 b1 = *(const float4*)(p + 20);
  v16bf r;
  r[0]=(__bf16)a0.x;  r[1]=(__bf16)a0.y;  r[2]=(__bf16)a0.z;  r[3]=(__bf16)a0.w;
  r[4]=(__bf16)a1.x;  r[5]=(__bf16)a1.y;  r[6]=(__bf16)a1.z;  r[7]=(__bf16)a1.w;
  r[8]=(__bf16)b0.x;  r[9]=(__bf16)b0.y;  r[10]=(__bf16)b0.z; r[11]=(__bf16)b0.w;
  r[12]=(__bf16)b1.x; r[13]=(__bf16)b1.y; r[14]=(__bf16)b1.z; r[15]=(__bf16)b1.w;
  return r;
}

#define WMMA_BF16(A, Bf, C) \
  __builtin_amdgcn_wmma_f32_16x16x32_bf16(false, (A), false, (Bf), (short)0, (C), false, false)

// ---------------- Kernel 1: weight conversion ----------------
__global__ void k_cvt_weights(const float* __restrict__ Vw,
                              const float* __restrict__ Ww,
                              __bf16* __restrict__ Vwb,
                              __bf16* __restrict__ Wwb) {
  int i = blockIdx.x * blockDim.x + threadIdx.x;
  const int N = 512 * 512;
  if (i < N) {
    Vwb[i] = (__bf16)Vw[i];
    Wwb[i] = (__bf16)Ww[i];
  }
}

// ---------------- Kernel 2: xV = x @ Vw^T + Vb ----------------
// M = S*B = 131072, K = I = 512, N = H = 512.
// Block: 256 threads (8 waves, 2 M x 4 N). Block tile M=32, N=512 -> x read once.
// Wave tile 16 x 128 (8 WMMA tiles). HBM-bound; wave oversubscription hides
// the per-wave B-fragment load latency.
__global__ __launch_bounds__(256) void k_xv(const float* __restrict__ x,
                                            const __bf16* __restrict__ Vwb,
                                            const float* __restrict__ Vb,
                                            __bf16* __restrict__ xV) {
  const int I = 512, H = 512;
  const int m0 = blockIdx.x * 32;
  const int wave = threadIdx.x >> 5, lane = threadIdx.x & 31;
  const int wm = wave >> 2, wn = wave & 3;
  const int r0 = m0 + wm * 16;        // wave's M base
  const int c0 = wn * 128;            // wave's N base
  const int lo = lane & 15, hi = lane >> 4;

  v8f acc[8];
#pragma unroll
  for (int tn = 0; tn < 8; ++tn) acc[tn] = (v8f)(0.0f);

  for (int kb = 0; kb < 16; ++kb) {
    const int k0 = kb * 32;
    // A fragment: x rows r0..r0+15, K block k0..k0+31, converted f32->bf16.
    const float* xp = x + (size_t)(r0 + lo) * I + k0 + hi * 8;
    v16bf af = cvtA(xp);
    // B fragments: 8 tiles of Vw rows (N), K contiguous.
    v16bf bfr[8];
#pragma unroll
    for (int tn = 0; tn < 8; ++tn) {
      const __bf16* wp = Vwb + (size_t)(c0 + tn * 16 + lo) * I + k0 + hi * 8;
      bfr[tn] = load_frag(wp, 16);
    }
#pragma unroll
    for (int tn = 0; tn < 8; ++tn) acc[tn] = WMMA_BF16(af, bfr[tn], acc[tn]);
  }

#pragma unroll
  for (int tn = 0; tn < 8; ++tn) {
    const int n = c0 + tn * 16 + lo;
    const float bias = Vb[n];
#pragma unroll
    for (int v = 0; v < 8; ++v) {
      const int m = r0 + v + 8 * hi;
      xV[(size_t)m * H + n] = (__bf16)(acc[tn][v] + bias);
    }
  }
}

// ---------------- Kernel 3: sequential recurrence ----------------
// 16 blocks x 1024 threads. Block g owns batch rows [16g, 16g+16).
// Wave w: tile = w&15 (16 output cols), kg = w>>4 (K half 0/1).
// Ww B-fragments resident in VGPRs (8 x v16bf per wave).
// h ping-pong in LDS (bf16, stride 520 halves -> conflict-free b128 reads).
// Symmetric K-split reduction: kg0 finalizes acc rows v=0..3, kg1 rows v=4..7;
// each parks the other group's rows in LDS. xV prefetched one step ahead.
__global__ __launch_bounds__(1024) void k_rnn(const __bf16* __restrict__ Wwb,
                                              const float* __restrict__ Wb,
                                              const __bf16* __restrict__ xV,
                                              float* __restrict__ out,
                                              float* __restrict__ out_last) {
  const int S = 512, B = 256, H = 512, LDH = 520;
  extern __shared__ char smem[];
  __bf16* hb   = (__bf16*)smem;                                  // [2][16][LDH]
  // partial buffer: word index = tile*288 + hi*144 + v*16 + lo
  // (144 mod 64 == 16 -> both 16-lane halves of a store hit disjoint banks)
  float*  part = (float*)(smem + 2 * 16 * LDH * sizeof(__bf16)); // 18432 B

  const int R = blockIdx.x * 16;                 // batch-row base
  const int tid = threadIdx.x;
  const int wave = tid >> 5, lane = tid & 31;
  const int tile = wave & 15, kg = wave >> 4;
  const int c0 = tile * 16;
  const int lo = lane & 15, hi = lane >> 4;
  const int vbase = kg * 4;       // acc rows this wave group finalizes
  const int obase = 4 - vbase;    // acc rows it parks for the other group

  // Preload this wave's Ww fragments: cols c0..c0+15, K in [256*kg, 256*kg+256).
  v16bf bfr[8];
#pragma unroll
  for (int kb = 0; kb < 8; ++kb) {
    const int k0 = kg * 256 + kb * 32;
    const __bf16* wp = Wwb + (size_t)(c0 + lo) * H + k0 + hi * 8;
    bfr[kb] = load_frag(wp, 16);
  }
  const float bias = Wb[c0 + lo];

  // h_0 = 0 in ping buffer 0.
  for (int i = tid; i < 16 * LDH; i += 1024) hb[i] = (__bf16)0.0f;
  __syncthreads();

  // Prefetch xV for t=0 (4 rows this lane finalizes).
  __bf16 xb[4];
#pragma unroll
  for (int j = 0; j < 4; ++j) {
    const int lr = vbase + j + 8 * hi;
    xb[j] = xV[((size_t)0 * B + R + lr) * H + c0 + lo];
  }

  for (int t = 0; t < S; ++t) {
    // Start prefetch of next step's xV; latency overlaps the WMMA chain.
    __bf16 xbn[4] = {};
    if (t + 1 < S) {
#pragma unroll
      for (int j = 0; j < 4; ++j) {
        const int lr = vbase + j + 8 * hi;
        xbn[j] = xV[((size_t)(t + 1) * B + R + lr) * H + c0 + lo];
      }
    }

    const __bf16* ping = hb + (t & 1) * (16 * LDH);
    __bf16* pong = hb + ((t + 1) & 1) * (16 * LDH);

    v8f acc = (v8f)(0.0f);
#pragma unroll
    for (int kb = 0; kb < 8; ++kb) {
      const int k = kg * 256 + kb * 32;
      const __bf16* ap = ping + lo * LDH + k + hi * 8;  // A: h rows = batch
      v16bf af = load_frag(ap, 16);
      acc = WMMA_BF16(af, bfr[kb], acc);
    }

    // Park the half of acc that the OTHER wave group finalizes.
#pragma unroll
    for (int j = 0; j < 4; ++j) {
      const int v = obase + j;
      part[tile * 288 + hi * 144 + v * 16 + lo] = acc[v];
    }
    __syncthreads();

    // Both groups: reduce own 4 rows, add xV+bias, tanh, emit h_t.
#pragma unroll
    for (int j = 0; j < 4; ++j) {
      const int v = vbase + j;
      const int lr = v + 8 * hi;
      const int row = R + lr;
      const size_t idx = ((size_t)t * B + row) * H + c0 + lo;
      float z = acc[v] + part[tile * 288 + hi * 144 + v * 16 + lo]
              + (float)xb[j] + bias;
      float hn = tanhf(z);
      out[idx] = hn;                                       // h_seq
      if (t == S - 1) out_last[(size_t)row * H + c0 + lo] = hn;
      pong[lr * LDH + c0 + lo] = (__bf16)hn;               // feedback
    }
    __syncthreads();

#pragma unroll
    for (int j = 0; j < 4; ++j) xb[j] = xbn[j];
  }
}

extern "C" void kernel_launch(void* const* d_in, const int* in_sizes, int n_in,
                              void* d_out, int out_size, void* d_ws, size_t ws_size,
                              hipStream_t stream) {
  (void)in_sizes; (void)n_in; (void)out_size; (void)ws_size;
  const float* x  = (const float*)d_in[0];
  const float* Vw = (const float*)d_in[1];
  const float* Vb = (const float*)d_in[2];
  const float* Ww = (const float*)d_in[3];
  const float* Wb = (const float*)d_in[4];

  char* ws = (char*)d_ws;
  __bf16* Vwb = (__bf16*)(ws);                 // 512*512*2 = 512 KB
  __bf16* Wwb = (__bf16*)(ws + 524288);        // 512 KB
  __bf16* xVb = (__bf16*)(ws + 1048576);       // 512*256*512*2 = 128 MB

  float* out = (float*)d_out;
  float* out_last = out + (size_t)512 * 256 * 512;

  k_cvt_weights<<<(512 * 512 + 255) / 256, 256, 0, stream>>>(Vw, Ww, Vwb, Wwb);
  k_xv<<<131072 / 32, 256, 0, stream>>>(x, Vwb, Vb, xVb);

  const size_t shmem = (size_t)2 * 16 * 520 * sizeof(__bf16)   // h ping-pong
                     + (size_t)16 * 288 * sizeof(float);       // skewed partials
  k_rnn<<<16, 1024, shmem, stream>>>(Wwb, Wb, xVb, out, out_last);
}
